// IPEXTransformerAtten_1365799600137
// MI455X (gfx1250) — matrix operand order repaired and checked by reading, not compiled, not run on hardware
//
#include <hip/hip_runtime.h>
#include <hip/hip_bf16.h>
#include <hip/hip_fp16.h>

// ---------------------------------------------------------------------------
// MI455X (gfx1250) attention block.  B=1, S=2048, E=4096, H=32, D=128.
// f16 WMMA (v_wmma_f32_16x16x32_f16) everywhere, fp32 accumulation.
// - GEMMs: 64x128 tile, software-pipelined global->regs->LDS staging,
//   pipelined LDS fragment loads.
// - Attention: 8-wave blocks (128 queries) sharing double-buffered async
//   global->LDS K/V tiles (ASYNCcnt), DPP row_xmask softmax reductions.
// ---------------------------------------------------------------------------

#define S_LEN 2048
#define EMB   4096
#define NH    32
#define HD    128

typedef __attribute__((ext_vector_type(16))) _Float16 v16h;
typedef __attribute__((ext_vector_type(8)))  float    v8f;
typedef __attribute__((ext_vector_type(4)))  _Float16 h4v;   // 8-byte packed

static __device__ inline v8f wmma_f16(v16h a, v16h b, v8f c) {
  return __builtin_amdgcn_wmma_f32_16x16x32_f16(
      /*neg_a=*/false, a, /*neg_b=*/false, b,
      /*c_mod=*/(short)0, c, /*reuse_a=*/false, /*reuse_b=*/false);
}

// Async 16B global -> LDS copy, per-lane addresses, tracked by ASYNCcnt.
static __device__ inline void async_load_b128(const _Float16* gptr,
                                              _Float16* lptr) {
  const unsigned int lds = (unsigned int)(uintptr_t)lptr;  // addr[31:0] = LDS offset
  asm volatile("global_load_async_to_lds_b128 %0, %1, off"
               :: "v"(lds), "v"(gptr)
               : "memory");
}
static __device__ inline void wait_async0() {
  asm volatile("s_wait_asynccnt 0" ::: "memory");
}
static __device__ inline void wait_async4() {   // current tile done, next in flight
  asm volatile("s_wait_asynccnt 4" ::: "memory");
}

// DPP16 ROW_XMASK cross-lane (xor within 16-lane row): pure VALU, no LDS.
template <int M>
static __device__ inline float dpp_xmask(float x) {
  return __int_as_float(__builtin_amdgcn_update_dpp(
      0, __float_as_int(x), 0x160 | M, 0xf, 0xf, true));
}
static __device__ inline float row16_max(float x) {
  x = fmaxf(x, dpp_xmask<1>(x));
  x = fmaxf(x, dpp_xmask<2>(x));
  x = fmaxf(x, dpp_xmask<4>(x));
  x = fmaxf(x, dpp_xmask<8>(x));
  return x;
}
static __device__ inline float row16_sum(float x) {
  x += dpp_xmask<1>(x);
  x += dpp_xmask<2>(x);
  x += dpp_xmask<4>(x);
  x += dpp_xmask<8>(x);
  return x;
}

// A-matrix 16x32 f16 fragment (ISA 7.12.2): lane m = lane&15,
// lanes 0-15 hold K = {0..7, 16..23}, lanes 16-31 hold K = {8..15, 24..31}.
static __device__ inline v16h load_a_frag(const _Float16* t, int stride) {
  const int lane = threadIdx.x & 31;
  const int m  = lane & 15;
  const int kb = (lane >> 4) << 3;            // 0 or 8
  const _Float16* p = t + m * stride + kb;
  union { v16h v; struct { float4 lo, hi; } f; } u;
  u.f.lo = *reinterpret_cast<const float4*>(p);        // K = kb+0..7
  u.f.hi = *reinterpret_cast<const float4*>(p + 16);   // K = kb+16..23
  return u.v;
}

// B-matrix 32x16 f16 fragment: lane n = lane&15, lanes 0-15 K={0..7,16..23},
// lanes 16-31 K={8..15,24..31}, source stored [n][k].
static __device__ inline v16h load_b_frag(const _Float16* t, int stride) {
  const int lane = threadIdx.x & 31;
  const int n  = lane & 15;
  const int kb = (lane >> 4) << 3;            // 0 or 8
  const _Float16* p = t + n * stride + kb;
  union { v16h v; struct { float4 lo, hi; } f; } u;
  u.f.lo = *reinterpret_cast<const float4*>(p);
  u.f.hi = *reinterpret_cast<const float4*>(p + 16);
  return u.v;
}

// ---------------------------------------------------------------------------
// NT GEMM: C[M,N] = A[M,K] * B[N,K]^T + bias, both operands K-contiguous.
// Block: 128 threads (4 waves), 64x128 tile, BK=32; wave w owns n-subtiles
// {2w,2w+1}: 8 WMMA per k-step per wave.  Global->register fetch of tile
// k+1 overlaps WMMA on tile k.
// ---------------------------------------------------------------------------
template <bool A_F16, bool C_F16>
__global__ __launch_bounds__(128)
void gemm_nt_kernel(const void* __restrict__ Aptr,
                    const float* __restrict__ Bw,
                    const float* __restrict__ bias,
                    void* __restrict__ Cptr,
                    int M, int N, int K) {
  __shared__ _Float16 As[64 * 40];    // 64 rows x 32 K, stride 40 halves
  __shared__ _Float16 Bs[128 * 40];   // 128 rows x 32 K

  const int tid  = threadIdx.x;
  const int lane = tid & 31;
  const int wave = tid >> 5;
  const int m0 = blockIdx.y * 64;
  const int n0 = blockIdx.x * 128;

  const float*    Af = (const float*)Aptr;
  const _Float16* Ah = (const _Float16*)Aptr;

  v8f acc[4][2];
#pragma unroll
  for (int i = 0; i < 4; i++)
#pragma unroll
    for (int j = 0; j < 2; j++) acc[i][j] = (v8f){0, 0, 0, 0, 0, 0, 0, 0};

  float4 ra[4];   // A staging (f16 path uses ra[0..1] as raw halves)
  float4 rb[8];   // B staging (fp32)

  auto fetch = [&](int kk) {
    if constexpr (A_F16) {
#pragma unroll
      for (int it = 0; it < 2; it++) {           // 256 8-half chunks
        const int c = tid + it * 128;
        const int row = c >> 2, c8 = c & 3;
        ra[it] = *reinterpret_cast<const float4*>(
            Ah + (size_t)(m0 + row) * K + kk + c8 * 8);
      }
    } else {
#pragma unroll
      for (int it = 0; it < 4; it++) {           // 512 4-float chunks
        const int c = tid + it * 128;
        const int row = c >> 3, c4 = c & 7;
        ra[it] = *reinterpret_cast<const float4*>(
            Af + (size_t)(m0 + row) * K + kk + c4 * 4);
      }
    }
#pragma unroll
    for (int it = 0; it < 8; it++) {             // 1024 4-float chunks
      const int c = tid + it * 128;
      const int row = c >> 3, c4 = c & 7;
      rb[it] = *reinterpret_cast<const float4*>(
          Bw + (size_t)(n0 + row) * K + kk + c4 * 4);
    }
  };

  auto store_tiles = [&]() {
    if constexpr (A_F16) {
#pragma unroll
      for (int it = 0; it < 2; it++) {
        const int c = tid + it * 128;
        const int row = c >> 2, c8 = c & 3;
        *reinterpret_cast<float4*>(&As[row * 40 + c8 * 8]) = ra[it];
      }
    } else {
#pragma unroll
      for (int it = 0; it < 4; it++) {
        const int c = tid + it * 128;
        const int row = c >> 3, c4 = c & 7;
        h4v ha;
        ha.x = (_Float16)ra[it].x; ha.y = (_Float16)ra[it].y;
        ha.z = (_Float16)ra[it].z; ha.w = (_Float16)ra[it].w;
        *reinterpret_cast<h4v*>(&As[row * 40 + c4 * 4]) = ha;
      }
    }
#pragma unroll
    for (int it = 0; it < 8; it++) {
      const int c = tid + it * 128;
      const int row = c >> 3, c4 = c & 7;
      h4v hb;
      hb.x = (_Float16)rb[it].x; hb.y = (_Float16)rb[it].y;
      hb.z = (_Float16)rb[it].z; hb.w = (_Float16)rb[it].w;
      *reinterpret_cast<h4v*>(&Bs[row * 40 + c4 * 4]) = hb;
    }
  };

  fetch(0);
  for (int kk = 0; kk < K; kk += 32) {
    __syncthreads();            // previous tile fully consumed
    store_tiles();
    __syncthreads();
    if (kk + 32 < K) fetch(kk + 32);   // overlaps the WMMAs below

    v16h b0 = load_b_frag(&Bs[(wave * 32 +  0) * 40], 40);
    v16h b1 = load_b_frag(&Bs[(wave * 32 + 16) * 40], 40);
    v16h af = load_a_frag(&As[0], 40);
#pragma unroll
    for (int i = 0; i < 4; i++) {
      v16h na;
      if (i < 3) na = load_a_frag(&As[((i + 1) * 16) * 40], 40);  // prefetch
      acc[i][0] = wmma_f16(af, b0, acc[i][0]);
      acc[i][1] = wmma_f16(af, b1, acc[i][1]);
      if (i < 3) af = na;
    }
  }

  // Epilogue: C layout — lane n = lane&15, rows r + 8*(lane>>4).
  const int n  = lane & 15;
  const int ro = (lane >> 4) << 3;
#pragma unroll
  for (int j = 0; j < 2; j++) {
    const int col = n0 + wave * 32 + j * 16 + n;
    const float bv = bias ? bias[col] : 0.0f;
#pragma unroll
    for (int i = 0; i < 4; i++) {
#pragma unroll
      for (int r = 0; r < 8; r++) {
        const int row = m0 + i * 16 + ro + r;
        const float v = acc[i][j][r] + bv;
        if constexpr (C_F16)
          ((_Float16*)Cptr)[(size_t)row * N + col] = (_Float16)v;
        else
          ((float*)Cptr)[(size_t)row * N + col] = v;
      }
    }
  }
}

// ---------------------------------------------------------------------------
// RoPE on q,k (f16 [s][e]) -> f16 packed [h][s][d].
// ---------------------------------------------------------------------------
__global__ void rope_pack_kernel(const _Float16* __restrict__ qh,
                                 const _Float16* __restrict__ kh,
                                 const int* __restrict__ pos,
                                 _Float16* __restrict__ qf,
                                 _Float16* __restrict__ kf) {
  const int i = blockIdx.x * blockDim.x + threadIdx.x;   // S*H*64
  if (i >= S_LEN * NH * 64) return;
  const int d = i & 63;
  const int h = (i >> 6) & (NH - 1);
  const int s = i >> 11;
  const float p = (float)pos[s];
  // inv_freq = 10000^(-d/64) = exp2(-d * log2(10000)/64)
  const float inv = exp2f(-(float)d * (13.287712379549449f / 64.0f));
  const float ang = p * inv;
  float sn, cs;
  __sincosf(ang, &sn, &cs);
  const size_t src = (size_t)s * EMB + (size_t)h * HD + d;
  const float q1 = (float)qh[src], q2 = (float)qh[src + 64];
  const float k1 = (float)kh[src], k2 = (float)kh[src + 64];
  const size_t dst = ((size_t)h * S_LEN + s) * HD + d;
  qf[dst]      = (_Float16)(q1 * cs - q2 * sn);
  qf[dst + 64] = (_Float16)(q2 * cs + q1 * sn);
  kf[dst]      = (_Float16)(k1 * cs - k2 * sn);
  kf[dst + 64] = (_Float16)(k2 * cs + k1 * sn);
}

// v [s][e] f16 -> vT [e][s] (= [h][d][s]) f16
__global__ void v_transpose_kernel(const _Float16* __restrict__ vh,
                                   _Float16* __restrict__ vT) {
  const int i = blockIdx.x * blockDim.x + threadIdx.x;
  if (i >= S_LEN * EMB) return;
  const int s = i & (S_LEN - 1);
  const int e = i >> 11;
  vT[(size_t)e * S_LEN + s] = vh[(size_t)s * EMB + e];
}

// ---------------------------------------------------------------------------
// Flash attention: block = 8 waves = 128 queries of one head (256 threads),
// sharing double-buffered async K/V tiles.  Per wave per 32-key block:
// 8 WMMAs scores + 8 WMMAs ctx, fragment loads software-pipelined.
// ---------------------------------------------------------------------------
__global__ __launch_bounds__(256)
void attn_kernel(const _Float16* __restrict__ qf,
                 const _Float16* __restrict__ kf,
                 const _Float16* __restrict__ vT,
                 _Float16* __restrict__ ctx) {
  __shared__ _Float16 Ks[2][32 * 136];   // 32 keys x 128 d (stride 136)
  __shared__ _Float16 Vs[2][128 * 40];   // 128 d x 32 keys (stride 40)
  __shared__ _Float16 Ps[8][16 * 40];    // per-wave 16 q x 32 k probs

  const int tid  = threadIdx.x;
  const int lane = tid & 31;
  const int wave = tid >> 5;
  const int h   = blockIdx.y;
  const int q0b = blockIdx.x * 128;
  const int q0w = q0b + wave * 16;

  const _Float16* kbase = kf + (size_t)h * S_LEN * HD;
  const _Float16* vbase = vT + (size_t)h * HD * S_LEN;

  auto issue_tile = [&](int j0, int buf) {
#pragma unroll
    for (int it = 0; it < 2; it++) {     // K: 512 16B chunks / 256 thr
      const int c = tid + it * 256;
      const int row = c >> 4, c8 = c & 15;
      async_load_b128(kbase + (size_t)(j0 + row) * HD + c8 * 8,
                      &Ks[buf][row * 136 + c8 * 8]);
    }
#pragma unroll
    for (int it = 0; it < 2; it++) {     // V: 512 16B chunks / 256 thr
      const int c = tid + it * 256;
      const int row = c >> 2, c8 = c & 3;
      async_load_b128(vbase + (size_t)row * S_LEN + j0 + c8 * 8,
                      &Vs[buf][row * 40 + c8 * 8]);
    }
  };

  // Query fragments, pre-scaled by 1/sqrt(D) (folds softmax scale into A).
  const _Float16 hisc = (_Float16)0.08838834764831845f;
  v16h aq[4];
#pragma unroll
  for (int kc = 0; kc < 4; kc++) {
    aq[kc] = load_a_frag(qf + ((size_t)h * S_LEN + q0w) * HD + kc * 32, HD);
    aq[kc] = aq[kc] * hisc;
  }

  float mx[8], l[8];
  v8f o[8];
#pragma unroll
  for (int r = 0; r < 8; r++) { mx[r] = -1e30f; l[r] = 0.0f; }
#pragma unroll
  for (int dt = 0; dt < 8; dt++) o[dt] = (v8f){0, 0, 0, 0, 0, 0, 0, 0};

  const int   n   = lane & 15;
  const int   ro  = (lane >> 4) << 3;
  const float L2E = 1.4426950408889634f;

  const int nkb = (q0b >> 5) + 4;        // 32-key blocks up to diagonal
  issue_tile(0, 0);                      // prologue
  for (int jb = 0; jb < nkb; jb++) {
    const int buf = jb & 1;
    const int j0  = jb * 32;
    if (jb + 1 < nkb) {                  // stream next tile behind compute
      issue_tile(j0 + 32, buf ^ 1);
      wait_async4();                     // in-order: current tile has landed
    } else {
      wait_async0();
    }
    __syncthreads();

    // --- scores: 16 q x 32 k, pipelined fragment loads ---
    v8f s0 = (v8f){0, 0, 0, 0, 0, 0, 0, 0};
    v8f s1 = (v8f){0, 0, 0, 0, 0, 0, 0, 0};
    v16h b0 = load_b_frag(&Ks[buf][0 * 136], 136);
    v16h b1 = load_b_frag(&Ks[buf][16 * 136], 136);
#pragma unroll
    for (int kc = 0; kc < 4; kc++) {
      v16h c0, c1;
      if (kc < 3) {
        c0 = load_b_frag(&Ks[buf][0  * 136 + (kc + 1) * 32], 136);
        c1 = load_b_frag(&Ks[buf][16 * 136 + (kc + 1) * 32], 136);
      }
      s0 = wmma_f16(aq[kc], b0, s0);
      s1 = wmma_f16(aq[kc], b1, s1);
      if (kc < 3) { b0 = c0; b1 = c1; }
    }

    // --- causal mask + online softmax (DPP row reductions) ---
    float p0[8], p1[8], fct[8];
#pragma unroll
    for (int r = 0; r < 8; r++) {
      const int q_idx = q0w + ro + r;
      const float a0 = (j0 + n      <= q_idx) ? s0[r] : -1e30f;
      const float a1 = (j0 + 16 + n <= q_idx) ? s1[r] : -1e30f;
      const float tm = row16_max(fmaxf(a0, a1));
      const float mn = fmaxf(mx[r], tm);
      const float f  = exp2f((mx[r] - mn) * L2E);
      p0[r] = exp2f((a0 - mn) * L2E);
      p1[r] = exp2f((a1 - mn) * L2E);
      const float rs = row16_sum(p0[r] + p1[r]);
      l[r]  = l[r] * f + rs;
      mx[r] = mn;
      fct[r] = f;
    }
#pragma unroll
    for (int dt = 0; dt < 8; dt++)
#pragma unroll
      for (int r = 0; r < 8; r++) o[dt][r] *= fct[r];

    // C-layout -> A-layout via per-wave LDS scratch.
    _Float16* ps = &Ps[wave][0];
#pragma unroll
    for (int r = 0; r < 8; r++) {
      const int m = ro + r;
      ps[m * 40 + n]      = (_Float16)p0[r];
      ps[m * 40 + 16 + n] = (_Float16)p1[r];
    }
    asm volatile("s_wait_dscnt 0" ::: "memory");   // same-wave LDS RAW
    v16h pa = load_a_frag(ps, 40);

    // --- ctx accumulation: 16 q x 128 d over K=32 keys, pipelined ---
    v16h bv = load_b_frag(&Vs[buf][0], 40);
#pragma unroll
    for (int dt = 0; dt < 8; dt++) {
      v16h nb;
      if (dt < 7) nb = load_b_frag(&Vs[buf][((dt + 1) * 16) * 40], 40);
      o[dt] = wmma_f16(pa, bv, o[dt]);
      if (dt < 7) bv = nb;
    }
    __syncthreads();   // all waves done with buf before it is refilled
  }

  // Normalize and write ctx in [s][e] f16 for the output projection.
#pragma unroll
  for (int r = 0; r < 8; r++) {
    const float inv = 1.0f / l[r];
    const int row = q0w + ro + r;
#pragma unroll
    for (int dt = 0; dt < 8; dt++) {
      ctx[(size_t)row * EMB + (size_t)h * HD + dt * 16 + n] =
          (_Float16)(o[dt][r] * inv);
    }
  }
}

// ---------------------------------------------------------------------------
// Host-side launch.
// ---------------------------------------------------------------------------
extern "C" void kernel_launch(void* const* d_in, const int* in_sizes, int n_in,
                              void* d_out, int out_size, void* d_ws, size_t ws_size,
                              hipStream_t stream) {
  const float* hs  = (const float*)d_in[0];
  const float* wq  = (const float*)d_in[1];
  const float* bq  = (const float*)d_in[2];
  const float* wk  = (const float*)d_in[3];
  const float* bk  = (const float*)d_in[4];
  const float* wv  = (const float*)d_in[5];
  const float* bvp = (const float*)d_in[6];
  const float* wo  = (const float*)d_in[7];
  const float* bo  = (const float*)d_in[8];
  const int*   pos = (const int*)d_in[9];
  float* out = (float*)d_out;

  char* ws = (char*)d_ws;
  const size_t HB = (size_t)S_LEN * EMB * sizeof(_Float16);   // 16 MB
  _Float16* qh  = (_Float16*)(ws + 0 * HB);   // q proj  [s][e]
  _Float16* kh  = (_Float16*)(ws + 1 * HB);   // k proj  [s][e]
  _Float16* vh  = (_Float16*)(ws + 2 * HB);   // v proj  [s][e]
  _Float16* qfp = (_Float16*)(ws + 3 * HB);   // q roped [h][s][d]
  _Float16* kfp = (_Float16*)(ws + 4 * HB);   // k roped [h][s][d]
  _Float16* vTp = (_Float16*)(ws + 5 * HB);   // v       [h][d][s]
  _Float16* ctx = (_Float16*)(ws + 6 * HB);   // context [s][e]

  const dim3 gg(EMB / 128, S_LEN / 64);
  gemm_nt_kernel<false, true><<<gg, 128, 0, stream>>>(hs, wq, bq,  qh, S_LEN, EMB, EMB);
  gemm_nt_kernel<false, true><<<gg, 128, 0, stream>>>(hs, wk, bk,  kh, S_LEN, EMB, EMB);
  gemm_nt_kernel<false, true><<<gg, 128, 0, stream>>>(hs, wv, bvp, vh, S_LEN, EMB, EMB);

  const int nrope = S_LEN * NH * 64;
  rope_pack_kernel<<<(nrope + 255) / 256, 256, 0, stream>>>(qh, kh, pos, qfp, kfp);
  const int ntr = S_LEN * EMB;
  v_transpose_kernel<<<(ntr + 255) / 256, 256, 0, stream>>>(vh, vTp);

  attn_kernel<<<dim3(S_LEN / 128, NH), 256, 0, stream>>>(qfp, kfp, vTp, ctx);

  gemm_nt_kernel<true, false><<<gg, 128, 0, stream>>>(ctx, wo, bo, out, S_LEN, EMB, EMB);
}
